// Model_51101520888080
// MI455X (gfx1250) — matrix operand order repaired and checked by reading, not compiled, not run on hardware
//
#include <hip/hip_runtime.h>
#include <hip/hip_bf16.h>
#include <math.h>

// ---------------------------------------------------------------------------
// MICN-style forecaster forward pass for MI455X (gfx1250, wave32, WMMA).
// All matmuls use v_wmma_f32_16x16x32_f16 (fp16 in / fp32 acc), register-
// blocked 4 N-tiles per wave (one A fragment feeds 4 WMMAs). The K-loop is
// left to the compiler to pipeline (explicit double-buffering caused spills).
// Input order (setup_inputs() dict insertion order, params flattened in
// insertion order):
//  0 x[32,96,321] f32   1 x_mark[32,96,4] i32   2 x_dec f32   3 x_mark_dec f32
//  4 start_W[32]  5 start_b[32]  6 tt_hour[24,32] 7 tt_weekday[7,32]
//  8 tt_day[32,32] 9 tt_month[13,32]
//  layer0: 10 conv_W[32,32,6] 11 conv_b 12 res_W[256,512] 13 res_b[256]
//  layer1: 14 conv_W[32,32,4] 15 conv_b 16 fr_W[321,321] 17 fr_b
//          18 fw_W[321,642] 19 fw_b 20 memory[4096,10272]
//          21 Wq 22 bq 23 Wk 24 bk 25 Wv 26 bv 27 mW[32,64] 28 mb
//          29 res_W[256,128] 30 res_b
//  layer2: 31..47 (same as layer1, res_W[256,32])
//  48 p1_W[512,256] 49 p1_b 50 p2_W[96,512] 51 p2_b 52 aw[321] 53 ab[321]
// Output: out[32,96,321] f32.
// ---------------------------------------------------------------------------

typedef __attribute__((ext_vector_type(16))) _Float16 v16h;
typedef __attribute__((ext_vector_type(8)))  _Float16 v8h;
typedef __attribute__((ext_vector_type(8)))  float    v8f;

#define BB 32
#define LL 96
#define CC 321
#define HH 32
#define PRED 96
#define NMEM 4096
#define ROWLEN (CC*HH)   // 10272
#define BC (BB*CC)       // 10272

// ---- WMMA fragment loaders (contiguous k = 16*(lane/16) + s in A and B) ------
__device__ __forceinline__ v16h load_afrag(const _Float16* A, int lda, int row,
                                           int k0, int lane) {
  const _Float16* p = A + (size_t)row * lda + k0 + ((lane >> 4) * 16);
  v8h lo = *(const v8h*)p;
  v8h hi = *(const v8h*)(p + 8);
  v16h a;
#pragma unroll
  for (int i = 0; i < 8; i++) { a[i] = lo[i]; a[8 + i] = hi[i]; }
  return a;
}

__device__ __forceinline__ v16h load_bfrag(const _Float16* Bw, int ldb, int col,
                                           int k0, int lane) {
  return *(const v16h*)(Bw + (size_t)col * ldb + k0 + ((lane >> 4) * 16));
}

// ---- Generic GEMM: Out[M,N] = A[M,K] * Bw[N,K]^T (+bias) ------------------
// Each wave computes a 16x64 tile: one A fragment feeds 4 WMMAs per K-step.
// Bw must be allocated (zero-padded) to nblk*64 rows. mode: 0=store,
// 1=accumulate into Out, 2=relu.
__global__ void k_gemm4(const _Float16* __restrict__ A, int lda,
                        const _Float16* __restrict__ Bw, int ldb,
                        float* __restrict__ Out, int ldo,
                        const float* __restrict__ bias,
                        int mtiles, int nblk, int N, int K, int mode) {
  int wave = blockIdx.x * (blockDim.x >> 5) + (threadIdx.x >> 5);
  if (wave >= mtiles * nblk) return;      // wave-uniform guard (EXEC all-1s)
  int tm = wave / nblk, tb = wave % nblk;
  int lane = threadIdx.x & 31;
  int rowA = tm * 16 + (lane & 15);
  int col0 = tb * 64 + (lane & 15);
  v8f acc[4] = {{}, {}, {}, {}};
  for (int k0 = 0; k0 < K; k0 += 32) {
    // branch-free prefetch of streamed B operand ~512B ahead (global_prefetch_b8)
    int kp = k0 + 256; kp = kp <= K - 32 ? kp : K - 32;
    __builtin_prefetch((const void*)(Bw + (size_t)col0 * ldb + kp), 0, 0);
    v16h af = load_afrag(A, lda, rowA, k0, lane);
    v16h b0 = load_bfrag(Bw, ldb, col0, k0, lane);
    v16h b1 = load_bfrag(Bw, ldb, col0 + 16, k0, lane);
    v16h b2 = load_bfrag(Bw, ldb, col0 + 32, k0, lane);
    v16h b3 = load_bfrag(Bw, ldb, col0 + 48, k0, lane);
    acc[0] = __builtin_amdgcn_wmma_f32_16x16x32_f16(false, af, false, b0,
                                                    (short)0, acc[0], false, false);
    acc[1] = __builtin_amdgcn_wmma_f32_16x16x32_f16(false, af, false, b1,
                                                    (short)0, acc[1], false, false);
    acc[2] = __builtin_amdgcn_wmma_f32_16x16x32_f16(false, af, false, b2,
                                                    (short)0, acc[2], false, false);
    acc[3] = __builtin_amdgcn_wmma_f32_16x16x32_f16(false, af, false, b3,
                                                    (short)0, acc[3], false, false);
  }
#pragma unroll
  for (int s = 0; s < 4; s++) {
    int col = tb * 64 + s * 16 + (lane & 15);
    if (col < N) {
      float bv = bias ? bias[col] : 0.0f;
#pragma unroll
      for (int i = 0; i < 8; i++) {
        int row = tm * 16 + i + ((lane >> 4) * 8);
        float v = acc[s][i] + bv;
        size_t o = (size_t)row * ldo + col;
        if (mode == 1) v += Out[o];
        if (mode == 2) v = v > 0.f ? v : 0.f;
        Out[o] = v;
      }
    }
  }
}

// ---- Strided conv1d as WMMA: per (b,c) tile [32 x cuts] -------------------
// Hin: [rows, T, 32] f16 ; Wc: [32, Kc] f16 with K = j*32 + h_in
__global__ void k_conv(const _Float16* __restrict__ Hin,
                       const _Float16* __restrict__ Wc,
                       const float* __restrict__ cb,
                       float* __restrict__ Y,   // [rows, 32, cuts]
                       int rows, int T, int cut, int cuts, int Kc) {
  int wave = blockIdx.x * (blockDim.x >> 5) + (threadIdx.x >> 5);
  if (wave >= rows * 2) return;
  int r = wave >> 1, mt = wave & 1;
  int lane = threadIdx.x & 31;
  int tcol = lane & 15;
  int kb16 = (lane >> 4) * 16;
  const _Float16* hbase = Hin + (size_t)r * T * 32;
  v8f acc = {};
  for (int k0 = 0; k0 < Kc; k0 += 32) {
    v16h af = load_afrag(Wc, Kc, mt * 16 + (lane & 15), k0, lane);
    v16h bf = {};
    if (tcol < cuts) {
      int j = k0 >> 5;   // K = j*32 + h_in ; run starts at h_in = kb16
      bf = *(const v16h*)(hbase + ((size_t)(tcol * cut + j) << 5) + kb16);
    }
    acc = __builtin_amdgcn_wmma_f32_16x16x32_f16(false, af, false, bf,
                                                 (short)0, acc, false, false);
  }
  if (tcol < cuts) {
#pragma unroll
    for (int i = 0; i < 8; i++) {
      int ho = mt * 16 + i + ((lane >> 4) * 8);
      Y[((size_t)r * 32 + ho) * cuts + tcol] = acc[i] + cb[ho];
    }
  }
}

// ---- Stats: mean/std over time per (b,c), one wave per row ----------------
__global__ void k_stats(const float* __restrict__ x, float* __restrict__ mean,
                        float* __restrict__ stdv) {
  int wave = blockIdx.x * (blockDim.x >> 5) + (threadIdx.x >> 5);
  if (wave >= BC) return;
  int b = wave / CC, c = wave % CC;
  int lane = threadIdx.x & 31;
  float s = 0.f, s2 = 0.f;
  for (int l = lane; l < LL; l += 32) {
    float v = x[((size_t)b * LL + l) * CC + c];
    s += v; s2 += v * v;
  }
  for (int o = 16; o; o >>= 1) { s += __shfl_xor(s, o, 32); s2 += __shfl_xor(s2, o, 32); }
  if (lane == 0) {
    float m = s / (float)LL;
    float var = s2 / (float)LL - m * m;
    mean[wave] = m;
    stdv[wave] = sqrtf(var + 1e-5f);
  }
}

// ---- Embedding: h16[b,c,l,h] = xn*start_W[h]+start_b[h]+temp[b,l,h] -------
__global__ void k_embed(const float* __restrict__ x, const int* __restrict__ xm,
                        const float* __restrict__ mean, const float* __restrict__ stdv,
                        const float* __restrict__ aw, const float* __restrict__ ab,
                        const float* __restrict__ sW, const float* __restrict__ sb,
                        const float* __restrict__ tth, const float* __restrict__ ttw,
                        const float* __restrict__ ttd, const float* __restrict__ ttm,
                        _Float16* __restrict__ h16) {
  int bl = blockIdx.x;            // b*96 + l
  int b = bl / LL, l = bl % LL;
  __shared__ float temp[HH], sWl[HH], sbl[HH];
  if (threadIdx.x < HH) {
    int h = threadIdx.x;
    int i0 = xm[bl * 4 + 0], i1 = xm[bl * 4 + 1], i2 = xm[bl * 4 + 2], i3 = xm[bl * 4 + 3];
    temp[h] = tth[i3 * HH + h] + ttw[i2 * HH + h] + ttd[i1 * HH + h] + ttm[i0 * HH + h];
    sWl[h] = sW[h]; sbl[h] = sb[h];
  }
  __syncthreads();
  for (int c = threadIdx.x; c < CC; c += blockDim.x) {
    float xv = x[(size_t)bl * CC + c];
    float m = mean[b * CC + c], sd = stdv[b * CC + c];
    float xn = (xv - m) / sd * aw[c] + ab[c];
    _Float16* dst = h16 + ((((size_t)b * CC + c) * LL) + l) * HH;
#pragma unroll
    for (int h = 0; h < HH; h++) dst[h] = (_Float16)(xn * sWl[h] + sbl[h] + temp[h]);
  }
}

// ---- Pack helpers ---------------------------------------------------------
__global__ void k_pack_plain(_Float16* __restrict__ dst, const float* __restrict__ src,
                             size_t n) {
  size_t i = (size_t)blockIdx.x * blockDim.x + threadIdx.x;
  if (i < n) dst[i] = (_Float16)src[i];
}

__global__ void k_pack_pad(_Float16* __restrict__ dst, const float* __restrict__ src,
                           int N, int K, int Npad, int Kpad, int maskdiag) {
  size_t i = (size_t)blockIdx.x * blockDim.x + threadIdx.x;
  if (i >= (size_t)Npad * Kpad) return;
  int n = (int)(i / Kpad), k = (int)(i % Kpad);
  float v = 0.f;
  if (n < N && k < K) { v = src[(size_t)n * K + k]; if (maskdiag && n == k) v = 0.f; }
  dst[i] = (_Float16)v;
}

__global__ void k_pack_fw(_Float16* __restrict__ dst, const float* __restrict__ fw) {
  // dst [384,672]; fw [321,642]; cat K-layout: [0,321)=xt, [336,657)=x1
  size_t i = (size_t)blockIdx.x * blockDim.x + threadIdx.x;
  if (i >= (size_t)384 * 672) return;
  int n = (int)(i / 672), k = (int)(i % 672);
  float v = 0.f;
  if (n < CC) {
    if (k < CC) v = fw[(size_t)n * 642 + k];
    else if (k >= 336 && k < 336 + CC) v = fw[(size_t)n * 642 + CC + (k - 336)];
  }
  dst[i] = (_Float16)v;
}

__global__ void k_pack_conv(_Float16* __restrict__ dst, const float* __restrict__ w,
                            int cut) {
  // dst [32, cut*32] with K = j*32+h_in ; w [hout, hin, cut]
  int Kc = cut * HH;
  int i = blockIdx.x * blockDim.x + threadIdx.x;
  if (i >= HH * Kc) return;
  int hout = i / Kc, kk = i % Kc, j = kk / HH, hin = kk % HH;
  dst[i] = (_Float16)w[((size_t)hout * HH + hin) * cut + j];
}

__global__ void k_xtpack(_Float16* __restrict__ cat, const float* __restrict__ Y,
                         int T) {
  // cat rows r = (b*32+h)*T + t ; first 336 slots (xt zero-padded)
  size_t i = (size_t)blockIdx.x * blockDim.x + threadIdx.x;
  size_t total = (size_t)BB * HH * T * 336;
  if (i >= total) return;
  int cc = (int)(i % 336); size_t r = i / 336;
  int t = (int)(r % T); size_t bh = r / T;
  int hh = (int)(bh % HH); int b = (int)(bh / HH);
  float v = (cc < CC) ? Y[((((size_t)b * CC) + cc) * HH + hh) * T + t] : 0.f;
  cat[r * 672 + cc] = (_Float16)v;
}

__global__ void k_x1pack(_Float16* __restrict__ cat, const float* __restrict__ x1,
                         size_t rows) {
  size_t i = (size_t)blockIdx.x * blockDim.x + threadIdx.x;
  if (i >= rows * 336) return;
  int cc = (int)(i % 336); size_t r = i / 336;
  float v = (cc < CC) ? x1[r * 336 + cc] : 0.f;
  cat[r * 672 + 336 + cc] = (_Float16)v;
}

__global__ void k_gating(_Float16* __restrict__ h16, const float* __restrict__ Y,
                         const float* __restrict__ x1, const float* __restrict__ bpre,
                         int T) {
  size_t i = (size_t)blockIdx.x * blockDim.x + threadIdx.x;
  size_t total = (size_t)BB * HH * T * CC;
  if (i >= total) return;
  int c = (int)(i % CC); size_t r = i / CC;
  int t = (int)(r % T); size_t bh = r / T;
  int hh = (int)(bh % HH); int b = (int)(bh / HH);
  float xt = Y[((((size_t)b * CC) + c) * HH + hh) * T + t];
  float xv = x1[r * 336 + c];
  float beta = 1.f / (1.f + __expf(-bpre[r * 336 + c]));
  float v = beta * xt + (1.f - beta) * xv;
  h16[((((size_t)b * CC) + c) * T + t) * HH + hh] = (_Float16)v;
}

// ---- Proto: q normalization, memory fp16 conversion, top-k, attention -----
__global__ void k_qnorm(const float* __restrict__ Y, _Float16* __restrict__ q16,
                        int T) {
  int n = blockIdx.x;           // b*T + t
  int b = n / T, t = n % T;
  __shared__ float red[256];
  float ss = 0.f;
  for (int i = threadIdx.x; i < ROWLEN; i += blockDim.x) {
    int c = i >> 5, h = i & 31;
    float v = Y[(((size_t)b * CC + c) * HH + h) * T + t];
    ss += v * v;
  }
  red[threadIdx.x] = ss; __syncthreads();
  for (int s2 = 128; s2; s2 >>= 1) {
    if ((int)threadIdx.x < s2) red[threadIdx.x] += red[threadIdx.x + s2];
    __syncthreads();
  }
  float inv = 1.f / fmaxf(sqrtf(red[0]), 1e-12f);
  for (int i = threadIdx.x; i < ROWLEN; i += blockDim.x) {
    int c = i >> 5, h = i & 31;
    float v = Y[(((size_t)b * CC + c) * HH + h) * T + t];
    q16[(size_t)n * ROWLEN + i] = (_Float16)(v * inv);
  }
}

__global__ void k_memconv(const float* __restrict__ mem, _Float16* __restrict__ mem16,
                          float* __restrict__ ss) {
  int r = blockIdx.x;
  __shared__ float red[256];
  float acc = 0.f;
  const float* src = mem + (size_t)r * ROWLEN;
  _Float16* dst = mem16 + (size_t)r * ROWLEN;
  for (int i = threadIdx.x; i < ROWLEN; i += blockDim.x) {
    float v = src[i]; dst[i] = (_Float16)v; acc += v * v;
  }
  red[threadIdx.x] = acc; __syncthreads();
  for (int s2 = 128; s2; s2 >>= 1) {
    if ((int)threadIdx.x < s2) red[threadIdx.x] += red[threadIdx.x + s2];
    __syncthreads();
  }
  if (threadIdx.x == 0) ss[r] = red[0];
}

__global__ void k_topk(const float* __restrict__ dots, const float* __restrict__ ss,
                       int* __restrict__ idxout) {
  int n = blockIdx.x;
  __shared__ float dbuf[NMEM];
  __shared__ float sval[256];
  __shared__ int sidx[256];
  for (int i = threadIdx.x; i < NMEM; i += blockDim.x)
    dbuf[i] = ss[i] - 2.f * dots[(size_t)n * NMEM + i];
  __syncthreads();
  for (int k = 0; k < 16; k++) {
    float bv = 3.4e38f; int bi = 0;
    for (int i = threadIdx.x; i < NMEM; i += blockDim.x) {
      float v = dbuf[i];
      if (v < bv) { bv = v; bi = i; }
    }
    sval[threadIdx.x] = bv; sidx[threadIdx.x] = bi; __syncthreads();
    for (int s2 = 128; s2; s2 >>= 1) {
      if ((int)threadIdx.x < s2 && sval[threadIdx.x + s2] < sval[threadIdx.x]) {
        sval[threadIdx.x] = sval[threadIdx.x + s2];
        sidx[threadIdx.x] = sidx[threadIdx.x + s2];
      }
      __syncthreads();
    }
    if (threadIdx.x == 0) { idxout[n * 16 + k] = sidx[0]; dbuf[sidx[0]] = 3.4e38f; }
    __syncthreads();
  }
}

// One wave per (n=(b,t), c); lane = h. Uses e_k = s_k.(Wk^T iq) + bk.iq and
// vh = Wv (sum a_k s_k) + bv (softmax weights sum to 1).
__global__ void k_attn(const float* __restrict__ X,     // [B*C, 32, T]
                       const float* __restrict__ mem,   // [4096, 10272] f32
                       const int* __restrict__ idx,     // [B*T, 16]
                       const float* __restrict__ Wq, const float* __restrict__ bq,
                       const float* __restrict__ Wk, const float* __restrict__ bk,
                       const float* __restrict__ Wv, const float* __restrict__ bv,
                       const float* __restrict__ mW, const float* __restrict__ mb,
                       float* __restrict__ X2, int T) {
  int wave = blockIdx.x * (blockDim.x >> 5) + (threadIdx.x >> 5);
  if (wave >= BB * T * CC) return;
  int c = wave % CC;
  int nt = wave / CC;              // b*T + t
  int b = nt / T, t = nt % T;
  int h = threadIdx.x & 31;
  size_t xoff = (((size_t)b * CC + c) * HH + h) * T + t;
  float x = X[xoff];
  float iq = bq[h];
#pragma unroll
  for (int j = 0; j < 32; j++) iq += Wq[h * 32 + j] * __shfl(x, j, 32);
  float u = 0.f;                   // u_j = (Wk^T iq)_j on lane j
#pragma unroll
  for (int j = 0; j < 32; j++) u += Wk[j * 32 + h] * __shfl(iq, j, 32);
  float bkiq = bk[h] * iq;
  for (int o = 16; o; o >>= 1) bkiq += __shfl_xor(bkiq, o, 32);
  const int* ip = idx + (size_t)nt * 16;
  float s[16], e[16];
  float emax = -3.4e38f;
#pragma unroll
  for (int k = 0; k < 16; k++) {
    int rr = ip[k];
    s[k] = mem[(size_t)rr * ROWLEN + c * 32 + h];
    float p = s[k] * u;
    for (int o = 16; o; o >>= 1) p += __shfl_xor(p, o, 32);
    e[k] = p + bkiq;
    emax = fmaxf(emax, e[k]);
  }
  float asum = 0.f, a[16];
#pragma unroll
  for (int k = 0; k < 16; k++) { a[k] = __expf(e[k] - emax); asum += a[k]; }
  float inv = 1.f / asum;
  float sbar = 0.f;
#pragma unroll
  for (int k = 0; k < 16; k++) sbar += a[k] * inv * s[k];
  float vh = bv[h];
#pragma unroll
  for (int j = 0; j < 32; j++) vh += Wv[h * 32 + j] * __shfl(sbar, j, 32);
  float pre = mb[h];
#pragma unroll
  for (int j = 0; j < 32; j++) pre += mW[h * 64 + j] * __shfl(x, j, 32);
#pragma unroll
  for (int j = 0; j < 32; j++) pre += mW[h * 64 + 32 + j] * __shfl(vh, j, 32);
  float alpha = 1.f / (1.f + __expf(-pre));
  X2[xoff] = x * alpha + vh * (1.f - alpha);
}

// ---- Final: transpose + de-normalize --------------------------------------
__global__ void k_final(const float* __restrict__ z,    // [B*C, 96]
                        const float* __restrict__ aw, const float* __restrict__ ab,
                        const float* __restrict__ mean, const float* __restrict__ stdv,
                        float* __restrict__ out) {
  size_t i = (size_t)blockIdx.x * blockDim.x + threadIdx.x;
  if (i >= (size_t)BB * PRED * CC) return;
  int c = (int)(i % CC);
  size_t r = i / CC;
  int t = (int)(r % PRED);
  int b = (int)(r / PRED);
  float v = z[(((size_t)b * CC) + c) * PRED + t];
  out[i] = (v - ab[c]) / (aw[c] + 1e-10f) * stdv[b * CC + c] + mean[b * CC + c];
}

// ---------------------------------------------------------------------------
static inline unsigned cdiv(size_t a, size_t b) { return (unsigned)((a + b - 1) / b); }

extern "C" void kernel_launch(void* const* d_in, const int* in_sizes, int n_in,
                              void* d_out, int out_size, void* d_ws, size_t ws_size,
                              hipStream_t stream) {
  (void)in_sizes; (void)n_in; (void)out_size; (void)ws_size;
  const float* x      = (const float*)d_in[0];
  const int*   xmark  = (const int*)d_in[1];
  const float* startW = (const float*)d_in[4];
  const float* startb = (const float*)d_in[5];
  const float* tth    = (const float*)d_in[6];
  const float* ttw    = (const float*)d_in[7];
  const float* ttd    = (const float*)d_in[8];
  const float* ttm    = (const float*)d_in[9];
  const float* p1W = (const float*)d_in[48]; const float* p1b = (const float*)d_in[49];
  const float* p2W = (const float*)d_in[50]; const float* p2b = (const float*)d_in[51];
  const float* aw  = (const float*)d_in[52]; const float* ab  = (const float*)d_in[53];

  // ---- workspace carving ----
  size_t off = 0;
  auto alloc = [&](size_t bytes) -> char* {
    char* p = (char*)d_ws + off;
    off += (bytes + 255) & ~(size_t)255;
    return p;
  };
  float*     mean  = (float*)alloc((size_t)BC * 4);
  float*     stdv  = (float*)alloc((size_t)BC * 4);
  _Float16*  h16   = (_Float16*)alloc((size_t)BC * LL * HH * 2);     // [B,C,L,H]
  float*     yA    = (float*)alloc((size_t)BC * HH * 16 * 4);        // conv/attn f32
  float*     yB    = (float*)alloc((size_t)BC * HH * 4 * 4);
  _Float16*  cat   = (_Float16*)alloc((size_t)16384 * 672 * 2);      // [xt | x1] f16
  float*     x1    = (float*)alloc((size_t)16384 * 336 * 4);
  float*     bpre  = (float*)alloc((size_t)16384 * 336 * 4);
  _Float16*  q16   = (_Float16*)alloc((size_t)128 * ROWLEN * 2);
  _Float16*  mem16 = (_Float16*)alloc((size_t)NMEM * ROWLEN * 2);    // 84 MB, L2-resident
  float*     memss = (float*)alloc((size_t)NMEM * 4);
  float*     dots  = (float*)alloc((size_t)128 * NMEM * 4);
  int*       idxb  = (int*)alloc((size_t)128 * 16 * 4);
  float*     skip  = (float*)alloc((size_t)BC * 256 * 4);
  _Float16*  a16   = (_Float16*)alloc((size_t)BC * 512 * 2);         // reused pack buf
  float*     z1    = (float*)alloc((size_t)BC * 512 * 4);
  _Float16*  frWp  = (_Float16*)alloc((size_t)384 * 352 * 2);        // 384 = 6*64 rows
  _Float16*  fwWp  = (_Float16*)alloc((size_t)384 * 672 * 2);
  _Float16*  cw16  = (_Float16*)alloc((size_t)32 * 192 * 2);
  _Float16*  rW16  = (_Float16*)alloc((size_t)256 * 512 * 2);
  _Float16*  pW16  = (_Float16*)alloc((size_t)512 * 256 * 2);        // also 128x512 p2
  float*     zbuf  = x1;   // reuse (head output [B*C,96] fits)

  const dim3 blk(256);
  // ntiles = ceil(N/16); B buffer must be zero-padded to ceil(ntiles/4)*64 rows
  auto gemm = [&](const _Float16* A, int lda, const _Float16* Bw, int ldb,
                  float* Out, int ldo, const float* bias,
                  int mtiles, int ntiles, int N, int K, int mode) {
    int nblk = (ntiles + 3) / 4;
    unsigned waves = (unsigned)(mtiles * nblk);
    hipLaunchKernelGGL(k_gemm4, dim3(cdiv(waves, 8)), blk, 0, stream,
                       A, lda, Bw, ldb, Out, ldo, bias, mtiles, nblk, N, K, mode);
  };

  // ---- stats + embedding ----
  hipLaunchKernelGGL(k_stats, dim3(cdiv(BC, 8)), blk, 0, stream, x, mean, stdv);
  hipLaunchKernelGGL(k_embed, dim3(BB * LL), blk, 0, stream, x, xmark, mean, stdv,
                     aw, ab, startW, startb, tth, ttw, ttd, ttm, h16);

  // ---- layer 0: conv(cut=6,cuts=16) + res ----
  hipLaunchKernelGGL(k_pack_conv, dim3(cdiv(32 * 192, 256)), blk, 0, stream,
                     cw16, (const float*)d_in[10], 6);
  hipLaunchKernelGGL(k_conv, dim3(cdiv(BC * 2, 8)), blk, 0, stream,
                     h16, cw16, (const float*)d_in[11], yA, BC, 96, 6, 16, 192);
  hipLaunchKernelGGL(k_pack_plain, dim3(cdiv((size_t)BC * 512, 256)), blk, 0, stream,
                     a16, yA, (size_t)BC * 512);
  hipLaunchKernelGGL(k_pack_plain, dim3(cdiv((size_t)256 * 512, 256)), blk, 0, stream,
                     rW16, (const float*)d_in[12], (size_t)256 * 512);
  gemm(a16, 512, rW16, 512, skip, 256, (const float*)d_in[13], BC / 16, 16, 256, 512, 0);

  // ---- layers 1 & 2 ----
  for (int L = 1; L <= 2; L++) {
    int base = (L == 1) ? 14 : 31;
    int Tin = (L == 1) ? 16 : 4;
    int cuts = (L == 1) ? 4 : 1;
    int Kres = cuts * HH;
    size_t rows = (size_t)BB * HH * Tin;          // interaction rows
    int nrows = BB * cuts;                        // proto rows (b,t)

    // interaction: x1 = xt @ (fr*mask)^T + fr_b ; beta = sigm(cat @ fw^T + fw_b)
    hipLaunchKernelGGL(k_xtpack, dim3(cdiv(rows * 336, 256)), blk, 0, stream, cat, yA, Tin);
    hipLaunchKernelGGL(k_pack_pad, dim3(cdiv((size_t)384 * 352, 256)), blk, 0, stream,
                       frWp, (const float*)d_in[base + 2], CC, CC, 384, 352, 1);
    gemm(cat, 672, frWp, 352, x1, 336, (const float*)d_in[base + 3],
         (int)(rows / 16), 21, CC, 352, 0);
    hipLaunchKernelGGL(k_x1pack, dim3(cdiv(rows * 336, 256)), blk, 0, stream, cat, x1, rows);
    hipLaunchKernelGGL(k_pack_fw, dim3(cdiv((size_t)384 * 672, 256)), blk, 0, stream,
                       fwWp, (const float*)d_in[base + 4]);
    gemm(cat, 672, fwWp, 672, bpre, 336, (const float*)d_in[base + 5],
         (int)(rows / 16), 21, CC, 672, 0);
    hipLaunchKernelGGL(k_gating, dim3(cdiv(rows * CC, 256)), blk, 0, stream,
                       h16, yA, x1, bpre, Tin);

    // conv (cut=4)
    hipLaunchKernelGGL(k_pack_conv, dim3(cdiv(32 * 128, 256)), blk, 0, stream,
                       cw16, (const float*)d_in[base + 0], 4);
    hipLaunchKernelGGL(k_conv, dim3(cdiv(BC * 2, 8)), blk, 0, stream,
                       h16, cw16, (const float*)d_in[base + 1], yB, BC, Tin, 4, cuts, 128);

    // proto: kNN over memory bank (WMMA distance GEMM against fp16 bank in L2)
    const float* memf = (const float*)d_in[base + 6];
    hipLaunchKernelGGL(k_qnorm, dim3(nrows), blk, 0, stream, yB, q16, cuts);
    hipLaunchKernelGGL(k_memconv, dim3(NMEM), blk, 0, stream, memf, mem16, memss);
    gemm(q16, ROWLEN, mem16, ROWLEN, dots, NMEM, nullptr,
         nrows / 16, NMEM / 16, NMEM, ROWLEN, 0);
    hipLaunchKernelGGL(k_topk, dim3(nrows), blk, 0, stream, dots, memss, idxb);
    hipLaunchKernelGGL(k_attn, dim3(cdiv((size_t)nrows * CC, 8)), blk, 0, stream,
                       yB, memf, idxb,
                       (const float*)d_in[base + 7],  (const float*)d_in[base + 8],
                       (const float*)d_in[base + 9],  (const float*)d_in[base + 10],
                       (const float*)d_in[base + 11], (const float*)d_in[base + 12],
                       (const float*)d_in[base + 13], (const float*)d_in[base + 14],
                       yA, cuts);

    // res accumulation
    hipLaunchKernelGGL(k_pack_plain, dim3(cdiv((size_t)BC * Kres, 256)), blk, 0, stream,
                       a16, yA, (size_t)BC * Kres);
    hipLaunchKernelGGL(k_pack_plain, dim3(cdiv((size_t)256 * Kres, 256)), blk, 0, stream,
                       rW16, (const float*)d_in[base + 15], (size_t)256 * Kres);
    gemm(a16, Kres, rW16, Kres, skip, 256, (const float*)d_in[base + 16],
         BC / 16, 16, 256, Kres, 1);
  }

  // ---- head: p1 (relu) -> p2 -> de-normalize ----
  hipLaunchKernelGGL(k_pack_plain, dim3(cdiv((size_t)BC * 256, 256)), blk, 0, stream,
                     a16, skip, (size_t)BC * 256);
  hipLaunchKernelGGL(k_pack_plain, dim3(cdiv((size_t)512 * 256, 256)), blk, 0, stream,
                     pW16, p1W, (size_t)512 * 256);
  gemm(a16, 256, pW16, 256, z1, 512, p1b, BC / 16, 32, 512, 256, 2);
  hipLaunchKernelGGL(k_pack_plain, dim3(cdiv((size_t)BC * 512, 256)), blk, 0, stream,
                     a16, z1, (size_t)BC * 512);
  // p2 B padded to 128 rows (nblk=2 => 2*64 rows)
  hipLaunchKernelGGL(k_pack_pad, dim3(cdiv((size_t)128 * 512, 256)), blk, 0, stream,
                     pW16, p2W, 96, 512, 128, 512, 0);
  gemm(a16, 512, pW16, 512, zbuf, 96, p2b, BC / 16, 6, 96, 512, 0);
  hipLaunchKernelGGL(k_final, dim3(cdiv((size_t)BB * PRED * CC, 256)), blk, 0, stream,
                     zbuf, aw, ab, mean, stdv, (float*)d_out);
}
